// flash_linear_trans_40922448396658
// MI455X (gfx1250) — matrix operand order repaired and checked by reading, not compiled, not run on hardware
//
#include <hip/hip_runtime.h>
#include <hip/hip_bf16.h>

// ---------------- problem constants ----------------
#define D_EMB   768
#define E_DIM   1536
#define S_DIM   128
#define NOUT    3200          // 2*E + S
#define LSEQ    4096
#define BATCH   8
#define MTOT    (BATCH * LSEQ)   // 32768
#define CHUNKC  256
#define NTILES  (BATCH * (LSEQ / CHUNKC))  // 128

// ---------------- workspace layout (bytes) ----------------
#define OFF_U      ((size_t)0)                                   // f16 [MTOT][E]  (becomes W in-place)
#define OFF_V      (OFF_U + (size_t)MTOT * E_DIM * 2)            // f16 [MTOT][E]  (reused as h f32 later)
#define OFF_BASE   (OFF_V + (size_t)MTOT * E_DIM * 2)            // f32 [MTOT][S]
#define OFF_ROLES  (OFF_BASE + (size_t)MTOT * S_DIM * 4)         // f16 [4][MTOT][S]
#define OFF_BIAS   (OFF_ROLES + (size_t)4 * MTOT * S_DIM * 2)    // f32 [256][256]

typedef _Float16 v16h __attribute__((ext_vector_type(16)));
typedef float    v8f  __attribute__((ext_vector_type(8)));

__device__ __forceinline__ v8f v8f_zero() {
  v8f z = {0.f,0.f,0.f,0.f,0.f,0.f,0.f,0.f};
  return z;
}

__device__ __forceinline__ v8f wmma_f16(v16h a, v16h b, v8f c) {
  // D = A(16x32 f16) * B(32x16 f16) + C(16x16 f32)
  return __builtin_amdgcn_wmma_f32_16x16x32_f16(
      /*neg_a=*/false, a, /*neg_b=*/false, b,
      /*c_mod=*/(short)0, c, /*reuse_a=*/false, /*reuse_b=*/false);
}

// fast SiLU: avoid the IEEE-division expansion (v_div_scale/fmas/fixup chain)
__device__ __forceinline__ float fast_silu(float v) {
  return v * __builtin_amdgcn_rcpf(1.f + __expf(-v));
}

// ---------------- async global->LDS copy (CDNA5 GLOBAL_LOAD_ASYNC_TO_LDS) ----------------
#if defined(__has_builtin)
# if __has_builtin(__builtin_amdgcn_global_load_async_to_lds_b128)
#  define HAVE_ASYNC_COPY 1
# endif
#endif
#ifndef HAVE_ASYNC_COPY
# define HAVE_ASYNC_COPY 0
#endif

typedef int async_i4 __attribute__((vector_size(16)));

__device__ __forceinline__ void cp_async16(void* ldst, const void* gsrc) {
#if HAVE_ASYNC_COPY
  __builtin_amdgcn_global_load_async_to_lds_b128(
      (__attribute__((address_space(1))) async_i4*)gsrc,
      (__attribute__((address_space(3))) async_i4*)ldst,
      /*imm offset=*/0, /*cpol=*/0);
#else
  *(uint4*)ldst = *(const uint4*)gsrc;
#endif
}

__device__ __forceinline__ void cp_async_wait() {
#if HAVE_ASYNC_COPY
# if defined(__has_builtin) && __has_builtin(__builtin_amdgcn_s_wait_asynccnt)
  __builtin_amdgcn_s_wait_asynccnt(0);
# else
  asm volatile("s_wait_asynccnt 0" ::: "memory");
# endif
#endif
}

// ---------------- WMMA fragment loaders (CDNA5 wave32 layouts) ----------------
// A fragment 16x32, element (m,k) at src[m*ld + k].
__device__ __forceinline__ v16h load_frag_a(const _Float16* src, int ld) {
  int lane = threadIdx.x & 31;
  int row  = lane & 15;
  int koff = (lane >> 4) * 16;
  const _Float16* p = src + row * ld + koff;
  v16h f;
#pragma unroll
  for (int i = 0; i < 8; ++i) f[i] = p[i];
#pragma unroll
  for (int i = 0; i < 8; ++i) f[8 + i] = p[8 + i];
  return f;
}

// A fragment 16x32, transposed storage: element (m,k) at srcT[k*ld + m].
__device__ __forceinline__ v16h load_frag_a_t(const _Float16* srcT, int ld) {
  int lane = threadIdx.x & 31;
  int row  = lane & 15;
  int koff = (lane >> 4) * 16;
  v16h f;
#pragma unroll
  for (int i = 0; i < 8; ++i) f[i]     = srcT[(koff + i) * ld + row];
#pragma unroll
  for (int i = 0; i < 8; ++i) f[8 + i] = srcT[(koff + 8 + i) * ld + row];
  return f;
}

// B fragment 32x16, transposed storage: element (k,n) at srcT[n*ld + k].
__device__ __forceinline__ v16h load_frag_b_t(const _Float16* srcT, int ld) {
  int lane = threadIdx.x & 31;
  int col  = lane & 15;
  int koff = (lane >> 4) * 16;
  const _Float16* p = srcT + col * ld + koff;
  v16h f;
#pragma unroll
  for (int i = 0; i < 16; ++i) f[i] = p[i];
  return f;
}

// B fragment 32x16, natural storage: element (k,n) at src[k*ld + n].
__device__ __forceinline__ v16h load_frag_b_n(const _Float16* src, int ld) {
  int lane = threadIdx.x & 31;
  int col  = lane & 15;
  int koff = (lane >> 4) * 16;
  v16h f;
#pragma unroll
  for (int i = 0; i < 16; ++i) f[i] = src[(koff + i) * ld + col];
  return f;
}

// =====================================================================
// Kernel 0: RoPE'd relative-position bias  bias[c][d] = a_c . b_d
// =====================================================================
__global__ __launch_bounds__(256)
void k_bias(const float* __restrict__ rel_a, const float* __restrict__ rel_b,
            float* __restrict__ bias) {
  __shared__ float sa[128];
  int c = blockIdx.x;          // row 0..255
  int t = threadIdx.x;
  if (t < 64) {
    float inv = __powf(10000.f, -(float)t / 64.f);
    float sn, cs; __sincosf((float)c * inv, &sn, &cs);
    float a1 = rel_a[t], a2 = rel_a[t + 64];
    sa[t]      = a1 * cs - a2 * sn;
    sa[t + 64] = a2 * cs + a1 * sn;
  }
  __syncthreads();
  int d = t;
  float dot = 0.f;
#pragma unroll 4
  for (int j = 0; j < 64; ++j) {
    float inv = __powf(10000.f, -(float)j / 64.f);
    float sn, cs; __sincosf((float)d * inv, &sn, &cs);
    float b1 = rel_b[j], b2 = rel_b[j + 64];
    dot += sa[j] * (b1 * cs - b2 * sn) + sa[j + 64] * (b2 * cs + b1 * sn);
  }
  bias[c * 256 + d] = dot;
}

// =====================================================================
// Kernel 1: projection GEMM  uvb = SiLU(x_shift @ UV_w^T + UV_b)
// block tile 128M x 128N, 8 waves (4Mx2N), wave tile 32x64, K step 64
// =====================================================================
__global__ __launch_bounds__(256)
void k_proj(const float* __restrict__ input, const float* __restrict__ UV_w,
            const float* __restrict__ UV_b,
            _Float16* __restrict__ Ubuf, _Float16* __restrict__ Vbuf,
            float* __restrict__ baseproj) {
  __shared__ _Float16 sA[128 * 64];
  __shared__ _Float16 sB[128 * 64];
  const int m0 = blockIdx.y * 128;
  const int n0 = blockIdx.x * 128;
  const int tid = threadIdx.x;
  const int wave = tid >> 5, lane = tid & 31;
  const int wm = (wave >> 1) * 32;   // 0..96
  const int wn = (wave & 1) * 64;    // 0,64

  v8f acc[2][4];
#pragma unroll
  for (int mi = 0; mi < 2; ++mi)
#pragma unroll
    for (int ni = 0; ni < 4; ++ni) acc[mi][ni] = v8f_zero();

  for (int kk = 0; kk < D_EMB; kk += 64) {
    __syncthreads();
    // stage A (token-shifted input, f32 -> f16)
    for (int i = tid; i < 8192; i += 256) {
      int r = i >> 6, c = i & 63;
      int m = m0 + r, k = kk + c;
      int l = m & (LSEQ - 1);
      float x;
      if (k < (D_EMB / 2)) x = (l == 0) ? 0.f : input[(size_t)(m - 1) * D_EMB + k];
      else                 x = input[(size_t)m * D_EMB + k];
      sA[r * 64 + c] = (_Float16)x;
    }
    // stage B: UV_w[n][k] (f32 -> f16), kept [n][k] for load_frag_b_t
    for (int i = tid; i < 8192; i += 256) {
      int r = i >> 6, c = i & 63;
      sB[r * 64 + c] = (_Float16)UV_w[(size_t)(n0 + r) * D_EMB + kk + c];
    }
    __syncthreads();
#pragma unroll
    for (int k2 = 0; k2 < 64; k2 += 32) {
      v16h a0 = load_frag_a(sA + (wm +  0) * 64 + k2, 64);
      v16h a1 = load_frag_a(sA + (wm + 16) * 64 + k2, 64);
#pragma unroll
      for (int ni = 0; ni < 4; ++ni) {
        v16h b = load_frag_b_t(sB + (wn + ni * 16) * 64 + k2, 64);
        acc[0][ni] = wmma_f16(a0, b, acc[0][ni]);
        acc[1][ni] = wmma_f16(a1, b, acc[1][ni]);
      }
    }
  }
  // epilogue: + bias, SiLU; destination is uniform per block (tile-aligned split)
  if (n0 < E_DIM) {                 // -> U
#pragma unroll
    for (int mi = 0; mi < 2; ++mi)
#pragma unroll
      for (int ni = 0; ni < 4; ++ni)
#pragma unroll
        for (int i = 0; i < 8; ++i) {
          int row = m0 + wm + mi * 16 + ((lane >> 4) << 3) + i;
          int col = n0 + wn + ni * 16 + (lane & 15);
          float s = fast_silu(acc[mi][ni][i] + UV_b[col]);
          Ubuf[(size_t)row * E_DIM + col] = (_Float16)s;
        }
  } else if (n0 < 2 * E_DIM) {      // -> V
#pragma unroll
    for (int mi = 0; mi < 2; ++mi)
#pragma unroll
      for (int ni = 0; ni < 4; ++ni)
#pragma unroll
        for (int i = 0; i < 8; ++i) {
          int row = m0 + wm + mi * 16 + ((lane >> 4) << 3) + i;
          int col = n0 + wn + ni * 16 + (lane & 15);
          float s = fast_silu(acc[mi][ni][i] + UV_b[col]);
          Vbuf[(size_t)row * E_DIM + (col - E_DIM)] = (_Float16)s;
        }
  } else {                          // -> base (f32)
#pragma unroll
    for (int mi = 0; mi < 2; ++mi)
#pragma unroll
      for (int ni = 0; ni < 4; ++ni)
#pragma unroll
        for (int i = 0; i < 8; ++i) {
          int row = m0 + wm + mi * 16 + ((lane >> 4) << 3) + i;
          int col = n0 + wn + ni * 16 + (lane & 15);
          float s = fast_silu(acc[mi][ni][i] + UV_b[col]);
          baseproj[(size_t)row * S_DIM + (col - 2 * E_DIM)] = s;
        }
  }
}

// =====================================================================
// Kernel 1b: per-role affine gate + RoPE -> 4 role tensors (f16)
// =====================================================================
__global__ __launch_bounds__(256)
void k_roles(const float* __restrict__ baseproj, const float* __restrict__ gamma,
             const float* __restrict__ beta, _Float16* __restrict__ roles) {
  int m = blockIdx.x;
  int t = threadIdx.x;
  int role = t >> 6;      // 0..3
  int j = t & 63;         // 0..63
  int pos = m & (LSEQ - 1);
  float inv = __powf(10000.f, -(float)j / 64.f);
  float sn, cs; __sincosf((float)pos * inv, &sn, &cs);
  float x1 = baseproj[(size_t)m * S_DIM + j];
  float x2 = baseproj[(size_t)m * S_DIM + j + 64];
  float g1 = x1 * gamma[role * S_DIM + j]      + beta[role * S_DIM + j];
  float g2 = x2 * gamma[role * S_DIM + j + 64] + beta[role * S_DIM + j + 64];
  _Float16* out = roles + (size_t)role * MTOT * S_DIM + (size_t)m * S_DIM;
  out[j]      = (_Float16)(g1 * cs - g2 * sn);
  out[j + 64] = (_Float16)(g2 * cs + g1 * sn);
}

// =====================================================================
// Kernel 2: fused chunk attention. One WG per (b,g) tile, 311 KB LDS.
// =====================================================================
#define K2_LDS (131072 + 65536 + 65536 + 32768 + 16384)   // 311296 B <= 320 KB
__global__ __launch_bounds__(256)
void k_attn(const _Float16* __restrict__ roles, const _Float16* __restrict__ Vbuf,
            _Float16* __restrict__ UW, const float* __restrict__ bias,
            float* __restrict__ prev) {
  extern __shared__ char smem[];
  _Float16* sP  = (_Float16*)smem;                       // [256][256] scores->probs
  _Float16* sQ  = (_Float16*)(smem + 131072);            // [256][128] quad_q, later lin_q
  _Float16* sK  = (_Float16*)(smem + 196608);            // [256][128] quad_k, later lin_k
  _Float16* sV  = (_Float16*)(smem + 262144);            // [256][64]  V tile
  _Float16* sKV = (_Float16*)(smem + 294912);            // [128][64]  lin_kv tile

  const int t = blockIdx.x;                    // 0..127
  const int tid = threadIdx.x;
  const int wave = tid >> 5, lane = tid & 31;
  const int m0w = wave * 32;
  const size_t rbase = (size_t)t * CHUNKC * S_DIM;
  const _Float16* qq = roles + 0 * (size_t)MTOT * S_DIM + rbase;
  const _Float16* qk = roles + 1 * (size_t)MTOT * S_DIM + rbase;
  const _Float16* lq = roles + 2 * (size_t)MTOT * S_DIM + rbase;
  const _Float16* lk = roles + 3 * (size_t)MTOT * S_DIM + rbase;

  // ---- stage quad_q / quad_k (async to LDS when available) ----
  {
    const uint4* gq = (const uint4*)qq;
    const uint4* gk = (const uint4*)qk;
    uint4* dq = (uint4*)sQ; uint4* dk = (uint4*)sK;
    for (int i = tid; i < 4096; i += 256) {
      cp_async16(dq + i, gq + i);
      cp_async16(dk + i, gk + i);
    }
    cp_async_wait();
  }
  __syncthreads();

  // ---- phase 1: scores = qq.qk^T / sqrt(S) + bias, write prev + sP ----
  const float scl = 0.0883883476483184f;  // 1/sqrt(128)
  for (int p = 0; p < 2; ++p) {           // two 128-col passes
    v8f acc[2][8];
#pragma unroll
    for (int mi = 0; mi < 2; ++mi)
#pragma unroll
      for (int ni = 0; ni < 8; ++ni) acc[mi][ni] = v8f_zero();
#pragma unroll
    for (int kk = 0; kk < S_DIM; kk += 32) {
      v16h a0 = load_frag_a(sQ + (m0w +  0) * S_DIM + kk, S_DIM);
      v16h a1 = load_frag_a(sQ + (m0w + 16) * S_DIM + kk, S_DIM);
#pragma unroll
      for (int ni = 0; ni < 8; ++ni) {
        v16h b = load_frag_b_t(sK + (p * 128 + ni * 16) * S_DIM + kk, S_DIM);
        acc[0][ni] = wmma_f16(a0, b, acc[0][ni]);
        acc[1][ni] = wmma_f16(a1, b, acc[1][ni]);
      }
    }
#pragma unroll
    for (int mi = 0; mi < 2; ++mi)
#pragma unroll
      for (int ni = 0; ni < 8; ++ni)
#pragma unroll
        for (int i = 0; i < 8; ++i) {
          int row = m0w + mi * 16 + ((lane >> 4) << 3) + i;
          int col = p * 128 + ni * 16 + (lane & 15);
          float raw = acc[mi][ni][i] * scl + bias[row * 256 + col];
          prev[(size_t)t * 65536 + row * 256 + col] = raw;
          sP[row * 256 + col] = (_Float16)raw;
        }
  }
  __syncthreads();

  // ---- phase 2: row softmax (one thread per row) ----
  {
    _Float16* rowp = sP + tid * 256;
    float mx = -3.0e38f;
    for (int j = 0; j < 256; ++j) mx = fmaxf(mx, (float)rowp[j]);
    float sum = 0.f;
    for (int j = 0; j < 256; ++j) sum += __expf((float)rowp[j] - mx);
    float invs = __builtin_amdgcn_rcpf(sum);
    for (int j = 0; j < 256; ++j)
      rowp[j] = (_Float16)(__expf((float)rowp[j] - mx) * invs);
  }
  __syncthreads();

  // ---- stage lin_q / lin_k over the Q/K slots ----
  {
    const uint4* glq = (const uint4*)lq;
    const uint4* glk = (const uint4*)lk;
    uint4* dq = (uint4*)sQ; uint4* dk = (uint4*)sK;
    for (int i = tid; i < 4096; i += 256) {
      cp_async16(dq + i, glq + i);
      cp_async16(dk + i, glk + i);
    }
    cp_async_wait();
  }
  __syncthreads();

  const float invL = 1.f / (float)LSEQ;
  // ---- phase 3: stream V in 64-wide tiles ----
  for (int e = 0; e < E_DIM / 64; ++e) {
    int ec0 = e * 64;
    for (int i = tid; i < 2048; i += 256) {   // stage V tile [256][64]
      int r = i >> 3, q = i & 7;
      cp_async16(((uint4*)sV) + i,
                 (const uint4*)(Vbuf + ((size_t)t * CHUNKC + r) * E_DIM + ec0 + q * 8));
    }
    cp_async_wait();
    __syncthreads();
    // lin_kv tile [128s][64]: wave owns 16 s-rows
    {
      v8f akv[4];
#pragma unroll
      for (int ni = 0; ni < 4; ++ni) akv[ni] = v8f_zero();
#pragma unroll
      for (int kk = 0; kk < 256; kk += 32) {
        v16h a = load_frag_a_t(sK + kk * S_DIM + wave * 16, S_DIM);
#pragma unroll
        for (int ni = 0; ni < 4; ++ni) {
          v16h b = load_frag_b_n(sV + kk * 64 + ni * 16, 64);
          akv[ni] = wmma_f16(a, b, akv[ni]);
        }
      }
#pragma unroll
      for (int ni = 0; ni < 4; ++ni)
#pragma unroll
        for (int i = 0; i < 8; ++i) {
          int row = wave * 16 + ((lane >> 4) << 3) + i;
          int col = ni * 16 + (lane & 15);
          sKV[row * 64 + col] = (_Float16)(akv[ni][i] * invL);
        }
    }
    __syncthreads();
    // quadratic + linear, combine with U, write W in place
    {
      v8f aq[2][4], al[2][4];
#pragma unroll
      for (int mi = 0; mi < 2; ++mi)
#pragma unroll
        for (int ni = 0; ni < 4; ++ni) { aq[mi][ni] = v8f_zero(); al[mi][ni] = v8f_zero(); }
#pragma unroll
      for (int kk = 0; kk < 256; kk += 32) {
        v16h a0 = load_frag_a(sP + (m0w +  0) * 256 + kk, 256);
        v16h a1 = load_frag_a(sP + (m0w + 16) * 256 + kk, 256);
#pragma unroll
        for (int ni = 0; ni < 4; ++ni) {
          v16h b = load_frag_b_n(sV + kk * 64 + ni * 16, 64);
          aq[0][ni] = wmma_f16(a0, b, aq[0][ni]);
          aq[1][ni] = wmma_f16(a1, b, aq[1][ni]);
        }
      }
#pragma unroll
      for (int kk = 0; kk < S_DIM; kk += 32) {
        v16h a0 = load_frag_a(sQ + (m0w +  0) * S_DIM + kk, S_DIM);
        v16h a1 = load_frag_a(sQ + (m0w + 16) * S_DIM + kk, S_DIM);
#pragma unroll
        for (int ni = 0; ni < 4; ++ni) {
          v16h b = load_frag_b_n(sKV + kk * 64 + ni * 16, 64);
          al[0][ni] = wmma_f16(a0, b, al[0][ni]);
          al[1][ni] = wmma_f16(a1, b, al[1][ni]);
        }
      }
#pragma unroll
      for (int mi = 0; mi < 2; ++mi)
#pragma unroll
        for (int ni = 0; ni < 4; ++ni)
#pragma unroll
          for (int i = 0; i < 8; ++i) {
            int row = m0w + mi * 16 + ((lane >> 4) << 3) + i;
            int col = ec0 + ni * 16 + (lane & 15);
            size_t gi = ((size_t)t * CHUNKC + row) * E_DIM + col;
            float u = (float)UW[gi];
            UW[gi] = (_Float16)(u * (aq[mi][ni][i] + al[mi][ni][i]));
          }
    }
    __syncthreads();
  }
}

// =====================================================================
// Kernel 3: output GEMM  h = W @ out_w^T + out_b + input (f32 into ws)
// block tile 128x128, K step 64
// =====================================================================
__global__ __launch_bounds__(256)
void k_out(const _Float16* __restrict__ Wbuf, const float* __restrict__ out_w,
           const float* __restrict__ out_b, const float* __restrict__ input,
           float* __restrict__ hbuf) {
  __shared__ _Float16 sA[128 * 64];
  __shared__ _Float16 sB[128 * 64];
  const int m0 = blockIdx.y * 128;
  const int n0 = blockIdx.x * 128;
  const int tid = threadIdx.x;
  const int wave = tid >> 5, lane = tid & 31;
  const int wm = (wave >> 1) * 32;
  const int wn = (wave & 1) * 64;

  v8f acc[2][4];
#pragma unroll
  for (int mi = 0; mi < 2; ++mi)
#pragma unroll
    for (int ni = 0; ni < 4; ++ni) acc[mi][ni] = v8f_zero();

  for (int kk = 0; kk < E_DIM; kk += 64) {
    __syncthreads();
    for (int i = tid; i < 1024; i += 256) {     // A: f16 copy [128][64] (async)
      int r = i >> 3, q = i & 7;
      cp_async16(((uint4*)sA) + i,
                 (const uint4*)(Wbuf + (size_t)(m0 + r) * E_DIM + kk + q * 8));
    }
    for (int i = tid; i < 8192; i += 256) {     // B: out_w[n][k] f32->f16
      int r = i >> 6, c = i & 63;
      sB[r * 64 + c] = (_Float16)out_w[(size_t)(n0 + r) * E_DIM + kk + c];
    }
    cp_async_wait();
    __syncthreads();
#pragma unroll
    for (int k2 = 0; k2 < 64; k2 += 32) {
      v16h a0 = load_frag_a(sA + (wm +  0) * 64 + k2, 64);
      v16h a1 = load_frag_a(sA + (wm + 16) * 64 + k2, 64);
#pragma unroll
      for (int ni = 0; ni < 4; ++ni) {
        v16h b = load_frag_b_t(sB + (wn + ni * 16) * 64 + k2, 64);
        acc[0][ni] = wmma_f16(a0, b, acc[0][ni]);
        acc[1][ni] = wmma_f16(a1, b, acc[1][ni]);
      }
    }
  }
#pragma unroll
  for (int mi = 0; mi < 2; ++mi)
#pragma unroll
    for (int ni = 0; ni < 4; ++ni)
#pragma unroll
      for (int i = 0; i < 8; ++i) {
        int row = m0 + wm + mi * 16 + ((lane >> 4) << 3) + i;
        int col = n0 + wn + ni * 16 + (lane & 15);
        float h = acc[mi][ni][i] + out_b[col] + input[(size_t)row * D_EMB + col];
        hbuf[(size_t)row * D_EMB + col] = h;
      }
}

// =====================================================================
// Kernel 4: LayerNorm over D_EMB=768, one block per row
// =====================================================================
__global__ __launch_bounds__(256)
void k_ln(const float* __restrict__ hbuf, const float* __restrict__ ln_g,
          const float* __restrict__ ln_b, float* __restrict__ y) {
  __shared__ float wsum[8], wssq[8];
  __shared__ float smu, srstd;
  int m = blockIdx.x;
  int tid = threadIdx.x;
  const float* hr = hbuf + (size_t)m * D_EMB;
  float v0 = hr[tid], v1 = hr[tid + 256], v2 = hr[tid + 512];
  float s = v0 + v1 + v2;
  float ss = v0 * v0 + v1 * v1 + v2 * v2;
#pragma unroll
  for (int o = 16; o > 0; o >>= 1) {
    s  += __shfl_down(s, o, 32);
    ss += __shfl_down(ss, o, 32);
  }
  if ((tid & 31) == 0) { wsum[tid >> 5] = s; wssq[tid >> 5] = ss; }
  __syncthreads();
  if (tid == 0) {
    float S = 0.f, SS = 0.f;
#pragma unroll
    for (int i = 0; i < 8; ++i) { S += wsum[i]; SS += wssq[i]; }
    float mu = S / (float)D_EMB;
    smu = mu;
    srstd = rsqrtf(SS / (float)D_EMB - mu * mu + 1e-5f);
  }
  __syncthreads();
  float mu = smu, r = srstd;
  float* yr = y + (size_t)m * D_EMB;
  yr[tid]       = (v0 - mu) * r * ln_g[tid]       + ln_b[tid];
  yr[tid + 256] = (v1 - mu) * r * ln_g[tid + 256] + ln_b[tid + 256];
  yr[tid + 512] = (v2 - mu) * r * ln_g[tid + 512] + ln_b[tid + 512];
}

// =====================================================================
extern "C" void kernel_launch(void* const* d_in, const int* in_sizes, int n_in,
                              void* d_out, int out_size, void* d_ws, size_t ws_size,
                              hipStream_t stream) {
  (void)in_sizes; (void)n_in; (void)out_size; (void)ws_size;
  const float* input = (const float*)d_in[0];
  const float* UV_w  = (const float*)d_in[1];
  const float* UV_b  = (const float*)d_in[2];
  const float* gamma = (const float*)d_in[3];
  const float* beta  = (const float*)d_in[4];
  const float* out_w = (const float*)d_in[5];
  const float* out_b = (const float*)d_in[6];
  const float* ln_g  = (const float*)d_in[7];
  const float* ln_b  = (const float*)d_in[8];
  const float* rel_a = (const float*)d_in[9];
  const float* rel_b = (const float*)d_in[10];

  char* ws = (char*)d_ws;
  _Float16* Ubuf    = (_Float16*)(ws + OFF_U);
  _Float16* Vbuf    = (_Float16*)(ws + OFF_V);
  float*    basepr  = (float*)(ws + OFF_BASE);
  _Float16* roles   = (_Float16*)(ws + OFF_ROLES);
  float*    biasbuf = (float*)(ws + OFF_BIAS);
  float*    hbuf    = (float*)(ws + OFF_V);   // reuse V region after k_attn

  float* y    = (float*)d_out;
  float* prev = y + (size_t)MTOT * D_EMB;     // outputs concatenated: y then prev

  k_bias <<<256, 256, 0, stream>>>(rel_a, rel_b, biasbuf);
  k_proj <<<dim3(NOUT / 128, MTOT / 128), 256, 0, stream>>>(
      input, UV_w, UV_b, Ubuf, Vbuf, basepr);
  k_roles<<<MTOT, 256, 0, stream>>>(basepr, gamma, beta, roles);
  k_attn <<<NTILES, 256, K2_LDS, stream>>>(roles, Vbuf, Ubuf, biasbuf, prev);
  k_out  <<<dim3(D_EMB / 128, MTOT / 128), 256, 0, stream>>>(
      Ubuf, out_w, out_b, input, hbuf);
  k_ln   <<<MTOT, 256, 0, stream>>>(hbuf, ln_g, ln_b, y);
}
